// InformerEncoderRegressor_43035572306471
// MI455X (gfx1250) — matrix operand order repaired and checked by reading, not compiled, not run on hardware
//
#include <hip/hip_runtime.h>
#include <hip/hip_bf16.h>
#include <cmath>

// ---------------------------------------------------------------------------
// Types for CDNA5 WMMA (gfx1250, wave32)
// ---------------------------------------------------------------------------
typedef _Float16 half_t;
typedef __attribute__((ext_vector_type(16))) _Float16 v16h;
typedef __attribute__((ext_vector_type(8)))  _Float16 v8h;
typedef __attribute__((ext_vector_type(8)))  float    v8f;

union AFrag { v16h v; v8h h[2]; };

#define GF_GELU 1
#define GF_RES  2

// ---------------------------------------------------------------------------
// Core WMMA GEMM:  out(M,N) = A(M,K) x W(N,K)^T  [+ bias][+ residual][GELU]
// A, W are f16 row-major. Block = 256 threads = 8 waves (4x2), block tile
// 128(M) x 64(N); each wave owns a 32x32 tile = 2x2 v_wmma_f32_16x16x32_f16.
// Requires M%128==0, N%64==0, K%32==0 (true for all call sites here).
// ---------------------------------------------------------------------------
__global__ __launch_bounds__(256)
void k_gemm_wmma(const half_t* __restrict__ A, const half_t* __restrict__ W,
                 const float* __restrict__ bias, const float* __restrict__ residual,
                 float* __restrict__ outF, half_t* __restrict__ outH,
                 int M, int N, int K, int flags)
{
    const int lane = threadIdx.x & 31;
    const int wave = threadIdx.x >> 5;        // 0..7
    const int wm   = wave >> 1;               // 0..3 (M dir)
    const int wn   = wave & 1;                // 0..1 (N dir)
    const int m0   = blockIdx.x * 128 + wm * 32;
    const int n0   = blockIdx.y * 64  + wn * 32;
    if (m0 >= M || n0 >= N) return;           // wave-uniform, never hit for our sizes

    v8f c00 = {}, c01 = {}, c10 = {}, c11 = {};

    // A fragment addressing (16-bit A 16x32 layout):
    // lane<16: elems 0..7 = K kh..kh+7, elems 8..15 = K 16+kh..; kh = (lane>>4)*8
    const int arow0 = m0 + (lane & 15);
    const int akh   = (lane >> 4) * 8;
    // B fragment addressing (16-bit B 32x16 layout): lane col = n&15,
    // elems 0..7 = K kh..kh+7, elems 8..15 = K 16+kh..; kh = (lane>>4)*8, from W row-major (N,K).
    const int bcol0 = n0 + (lane & 15);
    const int bko   = (lane >> 4) * 8;

    const half_t* a0p = A + (size_t)arow0        * K + akh;
    const half_t* a1p = A + (size_t)(arow0 + 16) * K + akh;
    const half_t* b0p = W + (size_t)bcol0        * K + bko;
    const half_t* b1p = W + (size_t)(bcol0 + 16) * K + bko;

    for (int kt = 0; kt < K; kt += 32) {
        __builtin_prefetch(a0p + kt + 128, 0, 1);   // global_prefetch_b8
        __builtin_prefetch(a1p + kt + 128, 0, 1);

        AFrag a0, a1, b0, b1;
        a0.h[0] = *(const v8h*)(a0p + kt);
        a0.h[1] = *(const v8h*)(a0p + kt + 16);
        a1.h[0] = *(const v8h*)(a1p + kt);
        a1.h[1] = *(const v8h*)(a1p + kt + 16);
        b0.h[0] = *(const v8h*)(b0p + kt);
        b0.h[1] = *(const v8h*)(b0p + kt + 16);
        b1.h[0] = *(const v8h*)(b1p + kt);
        b1.h[1] = *(const v8h*)(b1p + kt + 16);

        c00 = __builtin_amdgcn_wmma_f32_16x16x32_f16(false, a0.v, false, b0.v, (short)0, c00, false, false);
        c01 = __builtin_amdgcn_wmma_f32_16x16x32_f16(false, a0.v, false, b1.v, (short)0, c01, false, false);
        c10 = __builtin_amdgcn_wmma_f32_16x16x32_f16(false, a1.v, false, b0.v, (short)0, c10, false, false);
        c11 = __builtin_amdgcn_wmma_f32_16x16x32_f16(false, a1.v, false, b1.v, (short)0, c11, false, false);
    }

    // Epilogue: C/D layout -> lane col = n&15; VGPR r -> row r (lane<16) or r+8.
    auto store_tile = [&](v8f acc, int mt, int nt) {
        const int col = nt + (lane & 15);
        const float bv = bias ? bias[col] : 0.0f;
#pragma unroll
        for (int r = 0; r < 8; ++r) {
            const int row = mt + ((lane < 16) ? r : r + 8);
            const size_t idx = (size_t)row * N + col;
            float v = acc[r] + bv;
            if (flags & GF_RES)  v += residual[idx];
            if (flags & GF_GELU) v = 0.5f * v * (1.0f + erff(v * 0.70710678118f));
            if (outF) outF[idx] = v;
            if (outH) outH[idx] = (half_t)v;
        }
    };
    store_tile(c00, m0,      n0);
    store_tile(c10, m0 + 16, n0);
    store_tile(c01, m0,      n0 + 16);
    store_tile(c11, m0 + 16, n0 + 16);
}

// ---------------------------------------------------------------------------
// Small helper kernels (fp32 VALU paths)
// ---------------------------------------------------------------------------
__device__ inline float block_reduce_sum256(float v, float* red)
{
    int t = threadIdx.x;
    red[t] = v; __syncthreads();
    for (int o = 128; o > 0; o >>= 1) { if (t < o) red[t] += red[t + o]; __syncthreads(); }
    float r = red[0]; __syncthreads();
    return r;
}

__global__ void k_f32_to_f16(const float* __restrict__ src, half_t* __restrict__ dst, size_t n)
{
    size_t i = (size_t)blockIdx.x * blockDim.x + threadIdx.x;
    if (i < n) dst[i] = (half_t)src[i];
}

// circular-shift rows of h (B,L,D) and convert to f16 (conv A operand)
__global__ void k_shift_f16(const float* __restrict__ h, half_t* __restrict__ dst,
                            int L, int D, int shift, size_t n)
{
    size_t i = (size_t)blockIdx.x * blockDim.x + threadIdx.x;
    if (i >= n) return;
    int d  = (int)(i % D);
    size_t bl = i / D;
    int l  = (int)(bl % L);
    int b  = (int)(bl / L);
    int ls = (l + shift + L) % L;
    dst[i] = (half_t)h[((size_t)b * L + ls) * D + d];
}

// de-interleave dc_w: [E-1][Dout][Din][3] -> f16 [E-1][3][Dout][Din]
__global__ void k_dcw_deint(const float* __restrict__ src, half_t* __restrict__ dst,
                            int D, size_t n)
{
    size_t i = (size_t)blockIdx.x * blockDim.x + threadIdx.x;
    if (i >= n) return;
    int c  = (int)(i % D);
    int nn = (int)((i / D) % D);
    int kk = (int)((i / ((size_t)D * D)) % 3);
    int e  = (int)(i / ((size_t)3 * D * D));
    dst[i] = (half_t)src[(((size_t)e * D + nn) * D + c) * 3 + kk];
}

// skip = x[:, -1] @ skip_w.T + skip_b   (one thread per (b,p))
__global__ void k_skip(const float* __restrict__ x, const float* __restrict__ w,
                       const float* __restrict__ b, float* __restrict__ skip,
                       int B, int L, int Cin, int P)
{
    int id = blockIdx.x * blockDim.x + threadIdx.x;
    if (id >= B * P) return;
    int p = id % P, bb = id / P;
    float s = b[p];
    for (int c = 0; c < Cin; ++c)
        s += x[((size_t)bb * L + (L - 1)) * Cin + c] * w[p * Cin + c];
    skip[id] = s;
}

// token conv (circular, kernel 3, Cin->D) + positional embedding
__global__ __launch_bounds__(256)
void k_token_conv(const float* __restrict__ x, const float* __restrict__ w,
                  float* __restrict__ h, int L, int Cin, int D)
{
    int bl = blockIdx.x;
    int b = bl / L, l = bl % L;
    __shared__ float xs[3 * 8];
    if (threadIdx.x < 3 * Cin) {
        int kk = threadIdx.x / Cin, c = threadIdx.x % Cin;
        int ls = (l + kk - 1 + L) % L;
        xs[kk * Cin + c] = x[((size_t)b * L + ls) * Cin + c];
    }
    __syncthreads();
    for (int d = threadIdx.x; d < D; d += blockDim.x) {
        float acc = 0.0f;
        for (int kk = 0; kk < 3; ++kk)
            for (int c = 0; c < Cin; ++c)
                acc += xs[kk * Cin + c] * w[((size_t)d * Cin + c) * 3 + kk];
        // pos emb: i = d/2 ; f = exp(-2i*ln(1e4)/D)
        int i2 = d >> 1;
        float f = __expf(-(float)(2 * i2) * (9.210340372f / (float)D));
        float ang = (float)l * f;
        acc += (d & 1) ? __cosf(ang) : __sinf(ang);
        h[((size_t)b * L + l) * D + d] = acc;
    }
}

// row LayerNorm; optional f32 and f16 outputs
__global__ __launch_bounds__(256)
void k_layernorm(const float* __restrict__ x, const float* __restrict__ g,
                 const float* __restrict__ b, float* __restrict__ outF,
                 half_t* __restrict__ outH, int D)
{
    __shared__ float red[256];
    const float* xr = x + (size_t)blockIdx.x * D;
    float s = 0.0f;
    for (int d = threadIdx.x; d < D; d += 256) s += xr[d];
    float mu = block_reduce_sum256(s, red) / (float)D;
    float s2 = 0.0f;
    for (int d = threadIdx.x; d < D; d += 256) { float t = xr[d] - mu; s2 += t * t; }
    float rstd = rsqrtf(block_reduce_sum256(s2, red) / (float)D + 1e-5f);
    for (int d = threadIdx.x; d < D; d += 256) {
        float v = (xr[d] - mu) * rstd * g[d] + b[d];
        size_t idx = (size_t)blockIdx.x * D + d;
        if (outF) outF[idx] = v;
        if (outH) outH[idx] = (half_t)v;
    }
}

// prob-sparse sampling metric: m[b,h,l] = max_s qk - mean_s qk over u hashed keys
__global__ void k_qk_sample(const float* __restrict__ q, const float* __restrict__ k,
                            float* __restrict__ m, int B, int H, int L, int D,
                            int u, int seed)
{
    int id = blockIdx.x * blockDim.x + threadIdx.x;
    if (id >= B * H * L) return;
    int l = id % L; int bh = id / L; int hh = bh % H; int b = bh / H;
    const float* qr = q + ((size_t)b * L + l) * D + hh * 64;
    float qv[64];
#pragma unroll
    for (int d = 0; d < 64; ++d) qv[d] = qr[d];
    float mx = -1e30f, sum = 0.0f;
    for (int s = 0; s < u; ++s) {
        unsigned r = (unsigned)l * 2654435761u ^ (unsigned)s * 40503u ^ (unsigned)seed * 2246822519u;
        r ^= r >> 13; r *= 0x9E3779B1u; r ^= r >> 16;
        int idx = (int)(r % (unsigned)L);
        const float* kr = k + ((size_t)b * L + idx) * D + hh * 64;
        float acc = 0.0f;
#pragma unroll
        for (int d = 0; d < 64; ++d) acc += qv[d] * kr[d];
        mx = fmaxf(mx, acc); sum += acc;
    }
    m[id] = mx - sum / (float)u;
}

// per-(b,h) iterative top-u argmax over L (destroys m)
__global__ __launch_bounds__(256)
void k_topk(float* __restrict__ m, int* __restrict__ top, int L, int u)
{
    __shared__ float sv[256]; __shared__ int si[256];
    float* mr = m + (size_t)blockIdx.x * L;
    for (int j = 0; j < u; ++j) {
        float best = -1e30f; int bi = 0;
        for (int l = threadIdx.x; l < L; l += 256) {
            float v = mr[l];
            if (v > best) { best = v; bi = l; }
        }
        sv[threadIdx.x] = best; si[threadIdx.x] = bi; __syncthreads();
        for (int o = 128; o > 0; o >>= 1) {
            if (threadIdx.x < o && sv[threadIdx.x + o] > sv[threadIdx.x]) {
                sv[threadIdx.x] = sv[threadIdx.x + o];
                si[threadIdx.x] = si[threadIdx.x + o];
            }
            __syncthreads();
        }
        if (threadIdx.x == 0) { top[blockIdx.x * u + j] = si[0]; mr[si[0]] = -1e30f; }
        __syncthreads();
    }
}

// scores[b,h,j,:] = q_red . k / sqrt(dh)
__global__ __launch_bounds__(256)
void k_scores(const float* __restrict__ q, const float* __restrict__ k,
              const int* __restrict__ top, float* __restrict__ scores,
              int H, int L, int D, int u)
{
    __shared__ float qs[64];
    int bhj = blockIdx.x; int j = bhj % u; int bh = bhj / u;
    int hh = bh % H; int b = bh / H;
    int tq = top[bh * u + j];
    if (threadIdx.x < 64)
        qs[threadIdx.x] = q[((size_t)b * L + tq) * D + hh * 64 + threadIdx.x];
    __syncthreads();
    for (int l = threadIdx.x; l < L; l += 256) {
        const float* kr = k + ((size_t)b * L + l) * D + hh * 64;
        float acc = 0.0f;
#pragma unroll
        for (int d = 0; d < 64; ++d) acc += qs[d] * kr[d];
        scores[(size_t)bhj * L + l] = acc * 0.125f;  // 1/sqrt(64)
    }
}

__global__ __launch_bounds__(256)
void k_softmax(float* __restrict__ s, int L)
{
    __shared__ float red[256];
    float* r = s + (size_t)blockIdx.x * L;
    float mx = -1e30f;
    for (int l = threadIdx.x; l < L; l += 256) mx = fmaxf(mx, r[l]);
    red[threadIdx.x] = mx; __syncthreads();
    for (int o = 128; o > 0; o >>= 1) { if (threadIdx.x < o) red[threadIdx.x] = fmaxf(red[threadIdx.x], red[threadIdx.x + o]); __syncthreads(); }
    mx = red[0]; __syncthreads();
    float sum = 0.0f;
    for (int l = threadIdx.x; l < L; l += 256) { float e = __expf(r[l] - mx); r[l] = e; sum += e; }
    float tot = block_reduce_sum256(sum, red);
    float inv = 1.0f / tot;
    for (int l = threadIdx.x; l < L; l += 256) r[l] *= inv;
}

__global__ void k_vmean(const float* __restrict__ v, float* __restrict__ vmean, int L, int D)
{
    int b = blockIdx.x; int c = threadIdx.x;  // blockDim = D
    float s = 0.0f;
    for (int l = 0; l < L; ++l) s += v[((size_t)b * L + l) * D + c];
    vmean[b * D + c] = s / (float)L;
}

__global__ void k_ctx_fill(const float* __restrict__ vmean, float* __restrict__ ctx,
                           int L, int D, size_t n)
{
    size_t i = (size_t)blockIdx.x * blockDim.x + threadIdx.x;
    if (i >= n) return;
    int d = (int)(i % D);
    int b = (int)(i / ((size_t)L * D));
    ctx[i] = vmean[b * D + d];
}

__global__ void k_ctx_scatter(const float* __restrict__ attn, const float* __restrict__ v,
                              const int* __restrict__ top, float* __restrict__ ctx,
                              int H, int L, int D, int u)
{
    int bhj = blockIdx.x; int j = bhj % u; int bh = bhj / u;
    int hh = bh % H; int b = bh / H;
    int d = threadIdx.x;  // 64 threads
    const float* ar = attn + (size_t)bhj * L;
    float s = 0.0f;
    for (int l = 0; l < L; ++l) s += ar[l] * v[((size_t)b * L + l) * D + hh * 64 + d];
    ctx[((size_t)b * L + top[bh * u + j]) * D + hh * 64 + d] = s;
}

// batch-norm stats over (B,L) per channel
__global__ __launch_bounds__(256)
void k_bn_stats(const float* __restrict__ y, float* __restrict__ stats, int BL, int D)
{
    __shared__ float red[256];
    int c = blockIdx.x;
    float s = 0.0f, s2 = 0.0f;
    for (int r = threadIdx.x; r < BL; r += 256) {
        float t = y[(size_t)r * D + c];
        s += t; s2 += t * t;
    }
    float sum  = block_reduce_sum256(s,  red);
    float sum2 = block_reduce_sum256(s2, red);
    if (threadIdx.x == 0) {
        float mu = sum / (float)BL;
        stats[c] = mu;
        stats[D + c] = sum2 / (float)BL - mu * mu;
    }
}

// normalize + ELU + maxpool(k=3, stride=2, -inf pad) -> out (B, L/2, D)
__global__ void k_bn_elu_pool(const float* __restrict__ y, const float* __restrict__ stats,
                              const float* __restrict__ g, const float* __restrict__ beta,
                              float* __restrict__ out, int L, int D, size_t n)
{
    size_t i = (size_t)blockIdx.x * blockDim.x + threadIdx.x;
    if (i >= n) return;
    int Lo = L >> 1;
    int d  = (int)(i % D);
    size_t r = i / D;
    int io = (int)(r % Lo);
    int b  = (int)(r / Lo);
    float mu = stats[d];
    float rstd = rsqrtf(stats[D + d] + 1e-5f);
    float gd = g[d], bd = beta[d];
    float mx = -1e30f;
    for (int t = -1; t <= 1; ++t) {
        int l = 2 * io + t;
        if (l < 0 || l >= L) continue;
        float vv = (y[((size_t)b * L + l) * D + d] - mu) * rstd * gd + bd;
        vv = vv > 0.0f ? vv : (__expf(vv) - 1.0f);   // ELU
        mx = fmaxf(mx, vv);
    }
    out[((size_t)b * Lo + io) * D + d] = mx;
}

// final: LN(h[b,-1]) @ proj_w.T + proj_b + skip
__global__ __launch_bounds__(256)
void k_final(const float* __restrict__ h, const float* __restrict__ g,
             const float* __restrict__ b, const float* __restrict__ pw,
             const float* __restrict__ pb, const float* __restrict__ skip,
             float* __restrict__ out, int L, int D, int P)
{
    __shared__ float red[256];
    __shared__ float xn[512];
    int bb = blockIdx.x;
    const float* xr = h + ((size_t)bb * L + (L - 1)) * D;
    float s = 0.0f;
    for (int d = threadIdx.x; d < D; d += 256) s += xr[d];
    float mu = block_reduce_sum256(s, red) / (float)D;
    float s2 = 0.0f;
    for (int d = threadIdx.x; d < D; d += 256) { float t = xr[d] - mu; s2 += t * t; }
    float rstd = rsqrtf(block_reduce_sum256(s2, red) / (float)D + 1e-5f);
    for (int d = threadIdx.x; d < D; d += 256) xn[d] = (xr[d] - mu) * rstd * g[d] + b[d];
    __syncthreads();
    for (int p = threadIdx.x; p < P; p += 256) {
        float acc = pb[p];
        for (int d = 0; d < D; ++d) acc += xn[d] * pw[(size_t)p * D + d];
        out[bb * P + p] = acc + skip[bb * P + p];
    }
}

// ---------------------------------------------------------------------------
// Host orchestration
// ---------------------------------------------------------------------------
static inline size_t align256(size_t x) { return (x + 255) & ~(size_t)255; }
static inline int    cdiv(size_t a, int b) { return (int)((a + b - 1) / b); }

extern "C" void kernel_launch(void* const* d_in, const int* in_sizes, int n_in,
                              void* d_out, int out_size, void* d_ws, size_t ws_size,
                              hipStream_t stream)
{
    (void)in_sizes; (void)n_in; (void)out_size; (void)ws_size;
    constexpr int B = 32, L0 = 2048, CIN = 7, D = 512, DFF = 2048, E = 3, P = 96, H = 8;

    const float* x       = (const float*)d_in[0];
    const float* skip_w  = (const float*)d_in[1];
    const float* skip_b  = (const float*)d_in[2];
    const float* tok_w   = (const float*)d_in[3];
    const float* wq      = (const float*)d_in[4];
    const float* bq      = (const float*)d_in[5];
    const float* wk      = (const float*)d_in[6];
    const float* bk      = (const float*)d_in[7];
    const float* wv      = (const float*)d_in[8];
    const float* bv      = (const float*)d_in[9];
    const float* wo      = (const float*)d_in[10];
    const float* bo      = (const float*)d_in[11];
    const float* w1      = (const float*)d_in[12];
    const float* b1      = (const float*)d_in[13];
    const float* w2      = (const float*)d_in[14];
    const float* b2      = (const float*)d_in[15];
    const float* ln1_g   = (const float*)d_in[16];
    const float* ln1_b   = (const float*)d_in[17];
    const float* ln2_g   = (const float*)d_in[18];
    const float* ln2_b   = (const float*)d_in[19];
    const float* dc_w    = (const float*)d_in[20];
    const float* dc_b    = (const float*)d_in[21];
    const float* bn_g    = (const float*)d_in[22];
    const float* bn_b    = (const float*)d_in[23];
    const float* fln_g   = (const float*)d_in[24];
    const float* fln_b   = (const float*)d_in[25];
    const float* proj_w  = (const float*)d_in[26];
    const float* proj_b  = (const float*)d_in[27];
    float* out = (float*)d_out;

    // workspace carve
    char* base = (char*)d_ws; size_t off = 0;
    auto carve = [&](size_t bytes) -> void* { void* p = base + off; off = align256(off + bytes); return p; };
    float*  h     = (float*)carve((size_t)B * L0 * D * 4);
    float*  tmp   = (float*)carve((size_t)B * L0 * D * 4);   // ctx / FFN2-out
    float*  qb    = (float*)carve((size_t)B * L0 * D * 4);   // q / conv-y reuse
    float*  kb    = (float*)carve((size_t)B * L0 * D * 4);
    float*  vb    = (float*)carve((size_t)B * L0 * D * 4);
    half_t* a16   = (half_t*)carve((size_t)B * L0 * D * 2);  // shared f16 A operand
    half_t* y16   = (half_t*)carve((size_t)B * L0 * DFF * 2);
    float*  mbuf  = (float*)carve((size_t)B * H * L0 * 4);
    int*    top   = (int*)carve((size_t)B * H * 48 * 4);
    float*  scor  = (float*)carve((size_t)B * H * 48 * L0 * 4);
    float*  vmean = (float*)carve((size_t)B * D * 4);
    float*  stats = (float*)carve((size_t)2 * D * 4);
    float*  skipb = (float*)carve((size_t)B * P * 4);
    half_t* wq16  = (half_t*)carve((size_t)E * D * D * 2);
    half_t* wk16  = (half_t*)carve((size_t)E * D * D * 2);
    half_t* wv16  = (half_t*)carve((size_t)E * D * D * 2);
    half_t* wo16  = (half_t*)carve((size_t)E * D * D * 2);
    half_t* w116  = (half_t*)carve((size_t)E * DFF * D * 2);
    half_t* w216  = (half_t*)carve((size_t)E * DFF * D * 2);
    half_t* dcw16 = (half_t*)carve((size_t)(E - 1) * 3 * D * D * 2);

    auto cvt = [&](const float* s, half_t* d, size_t n) {
        k_f32_to_f16<<<cdiv(n, 256), 256, 0, stream>>>(s, d, n);
    };
    auto gemm = [&](const half_t* A, const half_t* W, const float* bias, const float* res,
                    float* oF, half_t* oH, int M, int N, int K, int flags) {
        dim3 g((M + 127) / 128, (N + 63) / 64);
        k_gemm_wmma<<<g, 256, 0, stream>>>(A, W, bias, res, oF, oH, M, N, K, flags);
    };

    // --- one-time weight conversion (deterministic, cheap) ---
    cvt(wq, wq16, (size_t)E * D * D);
    cvt(wk, wk16, (size_t)E * D * D);
    cvt(wv, wv16, (size_t)E * D * D);
    cvt(wo, wo16, (size_t)E * D * D);
    cvt(w1, w116, (size_t)E * DFF * D);
    cvt(w2, w216, (size_t)E * DFF * D);
    {
        size_t n = (size_t)(E - 1) * 3 * D * D;
        k_dcw_deint<<<cdiv(n, 256), 256, 0, stream>>>(dc_w, dcw16, D, n);
    }

    // --- skip path + stem ---
    k_skip<<<cdiv((size_t)B * P, 128), 128, 0, stream>>>(x, skip_w, skip_b, skipb, B, L0, CIN, P);
    k_token_conv<<<B * L0, 256, 0, stream>>>(x, tok_w, h, L0, CIN, D);

    // --- encoder layers ---
    int L = L0;
    for (int i = 0; i < E; ++i) {
        const int M = B * L;
        const int u = (int)(5.0 * log((double)L + 1.0));   // 38 / 34 / 31

        // QKV projections (WMMA)
        cvt(h, a16, (size_t)M * D);
        gemm(a16, wq16 + (size_t)i * D * D, bq + i * D, nullptr, qb, nullptr, M, D, D, 0);
        gemm(a16, wk16 + (size_t)i * D * D, bk + i * D, nullptr, kb, nullptr, M, D, D, 0);
        gemm(a16, wv16 + (size_t)i * D * D, bv + i * D, nullptr, vb, nullptr, M, D, D, 0);

        // prob-sparse attention
        k_qk_sample<<<cdiv((size_t)B * H * L, 256), 256, 0, stream>>>(qb, kb, mbuf, B, H, L, D, u, i);
        k_topk<<<B * H, 256, 0, stream>>>(mbuf, top, L, u);
        k_scores<<<B * H * u, 256, 0, stream>>>(qb, kb, top, scor, H, L, D, u);
        k_softmax<<<B * H * u, 256, 0, stream>>>(scor, L);
        k_vmean<<<B, D, 0, stream>>>(vb, vmean, L, D);
        k_ctx_fill<<<cdiv((size_t)M * D, 256), 256, 0, stream>>>(vmean, tmp, L, D, (size_t)M * D);
        k_ctx_scatter<<<B * H * u, 64, 0, stream>>>(scor, vb, top, tmp, H, L, D, u);

        // output projection with fused residual: h = h + ctx @ wo^T + bo
        cvt(tmp, a16, (size_t)M * D);
        gemm(a16, wo16 + (size_t)i * D * D, bo + i * D, h, h, nullptr, M, D, D, GF_RES);

        // FFN: y = gelu(LN1(h) @ w1^T + b1) ; h = LN2(h + y @ w2^T + b2)
        k_layernorm<<<M, 256, 0, stream>>>(h, ln1_g + i * D, ln1_b + i * D, nullptr, a16, D);
        gemm(a16, w116 + (size_t)i * DFF * D, b1 + i * DFF, nullptr, nullptr, y16, M, DFF, D, GF_GELU);
        gemm(y16, w216 + (size_t)i * D * DFF, b2 + i * D, h, tmp, nullptr, M, D, DFF, GF_RES);
        k_layernorm<<<M, 256, 0, stream>>>(tmp, ln2_g + i * D, ln2_b + i * D, h, nullptr, D);

        // distilling conv: circular 3-tap DxD conv as 3 shifted WMMA GEMMs,
        // then BN + ELU + maxpool(3, stride 2)
        if (i < E - 1) {
            for (int kk = 0; kk < 3; ++kk) {
                k_shift_f16<<<cdiv((size_t)M * D, 256), 256, 0, stream>>>(h, a16, L, D, kk - 1, (size_t)M * D);
                gemm(a16, dcw16 + (size_t)(i * 3 + kk) * D * D,
                     kk == 0 ? (dc_b + i * D) : nullptr,
                     kk == 0 ? nullptr : qb, qb, nullptr,
                     M, D, D, kk == 0 ? 0 : GF_RES);
            }
            k_bn_stats<<<D, 256, 0, stream>>>(qb, stats, M, D);
            size_t n = (size_t)B * (L / 2) * D;
            k_bn_elu_pool<<<cdiv(n, 256), 256, 0, stream>>>(qb, stats, bn_g + i * D, bn_b + i * D, h, L, D, n);
            L >>= 1;
        }
    }

    // final LN on last token + projection + skip
    k_final<<<B, 256, 0, stream>>>(h, fln_g, fln_b, proj_w, proj_b, skipb, out, L, D, P);
}
